// GraphNetFeature_16080357556246
// MI455X (gfx1250) — matrix operand order repaired and checked by reading, not compiled, not run on hardware
//
#include <hip/hip_runtime.h>

typedef __attribute__((ext_vector_type(16))) _Float16 v16h;
typedef __attribute__((ext_vector_type(8)))  float    v8f;

#define C 128
#define EPSV 1e-5f

// ---------------- init: deg = 1.0 (self loop), stats = 0 ----------------
__global__ void k_init(float* __restrict__ deg, float* __restrict__ stats, int n) {
    int t = blockIdx.x * blockDim.x + threadIdx.x;
    if (t < n) deg[t] = 1.0f;
    if (t < 768) stats[t] = 0.0f;
}

// ---------------- degree accumulation over dst ----------------
__global__ void k_deg(const long long* __restrict__ dst, float* __restrict__ deg, int E) {
    int t = blockIdx.x * blockDim.x + threadIdx.x;
    if (t < E) unsafeAtomicAdd(&deg[(int)dst[t]], 1.0f);
}

// ---------------- dinv = rsqrt(deg), in place ----------------
__global__ void k_dinv(float* __restrict__ deg, int n) {
    int t = blockIdx.x * blockDim.x + threadIdx.x;
    if (t < n) deg[t] = rsqrtf(deg[t]);
}

// ---------------- column stats (sum, sumsq) of [n,128] f32 ----------------
__global__ void k_colstats(const float* __restrict__ x, float* __restrict__ sum,
                           float* __restrict__ sumsq, int n, int rpb) {
    int c  = threadIdx.x;                 // blockDim = 128
    int r0 = blockIdx.x * rpb;
    int r1 = min(n, r0 + rpb);
    float s = 0.f, s2 = 0.f;
    for (int r = r0; r < r1; ++r) {
        float v = x[(size_t)r * C + c];
        s += v; s2 += v * v;
    }
    unsafeAtomicAdd(&sum[c], s);
    unsafeAtomicAdd(&sumsq[c], s2);
}

// ---------------- finalize: sum->mean, sumsq->rstd ----------------
__global__ void k_finalize(float* __restrict__ sum, float* __restrict__ sumsq, float n) {
    int c = threadIdx.x;                  // 1 block, 128 threads
    float mean = sum[c] / n;
    float var  = sumsq[c] / n - mean * mean;
    sum[c]   = mean;
    sumsq[c] = rsqrtf(var + EPSV);
}

// ---------------- BN apply, emit f16 GEMM operand ----------------
__global__ void k_bn_apply_f16(const float* __restrict__ x, const float* __restrict__ mean,
                               const float* __restrict__ rstd, const float* __restrict__ gamma,
                               const float* __restrict__ beta, _Float16* __restrict__ out,
                               int total) {
    int t = blockIdx.x * blockDim.x + threadIdx.x;
    if (t >= total) return;
    int c = t & (C - 1);
    out[t] = (_Float16)(gamma[c] * (x[t] - mean[c]) * rstd[c] + beta[c]);
}

// ---------------- f32 -> f16 weight convert ----------------
__global__ void k_f32_to_f16(const float* __restrict__ w, _Float16* __restrict__ wh, int total) {
    int t = blockIdx.x * blockDim.x + threadIdx.x;
    if (t < total) wh[t] = (_Float16)w[t];
}

// ---------------- WMMA GEMM + fused self-loop ----------------
// D[n,128] = A[n,128](f16) x W[128,128](f16), f32 acc.
// Also writes Agg = D * dinv[row]^2 (GCN self-loop term) from registers,
// saving a full 51MB read pass per layer.
// 256 threads = 8 waves; wave wv owns output columns [16wv, 16wv+16).
// W cached TRANSPOSED in LDS so B fragments are contiguous -> ds_load_b128.
#define LDSTRIDE 136
__global__ __launch_bounds__(256) void k_gemm_wmma(const _Float16* __restrict__ A,
                                                   const _Float16* __restrict__ Wh,
                                                   float* __restrict__ Dout,
                                                   float* __restrict__ Agg,
                                                   const float* __restrict__ dinv,
                                                   int n, int tilesPerBlock) {
    __shared__ _Float16 Wt[C * LDSTRIDE];   // transposed: Wt[n*LDSTRIDE + k]
    __shared__ _Float16 Al[16 * LDSTRIDE];  // A tile, row-major: Al[m*LDSTRIDE + k]
    const int tid  = threadIdx.x;
    const int lane = tid & 31;
    const int wv   = tid >> 5;
    const int m    = lane & 15;       // row within A-frag / col within B,C,D-frag
    const int hi   = lane >> 4;       // lane half select

    for (int i = tid; i < C * C; i += 256) {
        int k = i >> 7, nn = i & (C - 1);
        Wt[nn * LDSTRIDE + k] = Wh[i];  // coalesced global read, transposed LDS store
    }
    __syncthreads();

    const int tile0 = blockIdx.x * tilesPerBlock;
    const int col   = wv * 16 + m;
    // per-lane fragment base addresses in LDS (in halves)
    const int aBase = m * LDSTRIDE + (hi << 3);             // + kc*32 (+16 upper run)
    const int bBase = col * LDSTRIDE + (hi << 3);           // + kc*32 (+16 upper run)

    for (int t = 0; t < tilesPerBlock; ++t) {
        const int row0 = (tile0 + t) * 16;
        if (row0 >= n) break;
        const bool full = (row0 + 16 <= n);

        if (full) {
            const _Float16* Ab = A + (size_t)row0 * C;
            for (int i = tid; i < 16 * C; i += 256)
                Al[(i >> 7) * LDSTRIDE + (i & (C - 1))] = Ab[i];
        } else {
            for (int i = tid; i < 16 * C; i += 256) {
                int r = i >> 7, cc = i & (C - 1);
                int rr = row0 + r;
                Al[r * LDSTRIDE + cc] = (rr < n) ? A[(size_t)rr * C + cc] : (_Float16)0.f;
            }
        }
        // prefetch next A tile into cache while we compute on this one
        {
            int nrow = row0 + 16 + (tid >> 4);   // 256 threads cover 16 rows x 16 chunks
            if (nrow < n)
                __builtin_prefetch(A + (size_t)nrow * C + ((tid & 15) << 3), 0, 3);
        }
        __syncthreads();

        v8f acc = {};
        #pragma unroll
        for (int kc = 0; kc < 4; ++kc) {          // K = 128 in 4 chunks of 32
            v16h a, b;
            #pragma unroll
            for (int i = 0; i < 8; ++i) {
                // A 16-bit 16x32: lanes 0-15 K={0..7,16..23}, lanes 16-31 K={8..15,24..31}
                a[i]     = Al[aBase + kc * 32 + i];        // contiguous 8 halves
                a[i + 8] = Al[aBase + kc * 32 + 16 + i];   // contiguous 8 halves
            }
            #pragma unroll
            for (int i = 0; i < 8; ++i) {
                // B 16-bit 32x16: lane = N, lanes 0-15 K={0..7,16..23}, lanes 16-31 K={8..15,24..31}
                b[i]     = Wt[bBase + kc * 32 + i];        // contiguous 8 halves
                b[i + 8] = Wt[bBase + kc * 32 + 16 + i];   // contiguous 8 halves
            }
            acc = __builtin_amdgcn_wmma_f32_16x16x32_f16(
                    false, a, false, b, (short)0, acc, false, false);
        }

        // C/D layout: VGPR r -> M = r + 8*hi, N = lane&15
        const int rbase = row0 + (hi << 3);
        if (full) {
            // one base address, constant r*C offsets -> immediate-offset stores
            float*       dp = Dout + (size_t)rbase * C + col;
            float*       ap = Agg  + (size_t)rbase * C + col;
            const float* ip = dinv + rbase;
            #pragma unroll
            for (int r = 0; r < 8; ++r) {
                float v  = acc[r];
                float di = ip[r];
                dp[r * C] = v;
                ap[r * C] = v * di * di;
            }
        } else {
            #pragma unroll
            for (int r = 0; r < 8; ++r) {
                int row = rbase + r;
                if (row < n) {
                    float v  = acc[r];
                    float di = dinv[row];
                    Dout[(size_t)row * C + col] = v;
                    Agg[(size_t)row * C + col]  = v * di * di;
                }
            }
        }
        __syncthreads();
    }
}

// ---------------- edge scatter: agg[d] += h[s] * dinv[s]*dinv[d] ----------------
__global__ void k_edge(const long long* __restrict__ src, const long long* __restrict__ dst,
                       const float* __restrict__ h, const float* __restrict__ dinv,
                       float* __restrict__ agg, long long total) {
    long long t = (long long)blockIdx.x * blockDim.x + threadIdx.x;
    if (t >= total) return;
    int e = (int)(t >> 7);
    int c = (int)(t & (C - 1));
    int s = (int)src[e];
    int d = (int)dst[e];
    float nrm = dinv[s] * dinv[d];
    unsafeAtomicAdd(&agg[(size_t)d * C + c], h[(size_t)s * C + c] * nrm);
}

// ---------------- epilogue: z = relu(agg + bias) in place + column stats ----------------
__global__ void k_epi_stats(float* __restrict__ agg, const float* __restrict__ bias,
                            float* __restrict__ sum, float* __restrict__ sumsq,
                            int n, int rpb) {
    int c  = threadIdx.x;                 // blockDim = 128
    int r0 = blockIdx.x * rpb;
    int r1 = min(n, r0 + rpb);
    float b = bias[c];
    float s = 0.f, s2 = 0.f;
    for (int r = r0; r < r1; ++r) {
        size_t idx = (size_t)r * C + c;
        float v = agg[idx] + b;
        v = v > 0.f ? v : 0.f;
        agg[idx] = v;
        s += v; s2 += v * v;
    }
    unsafeAtomicAdd(&sum[c], s);
    unsafeAtomicAdd(&sumsq[c], s2);
}

// ---------------- final: out = bn0(x) + bn2(z2) (x_norm recomputed, saves a pass) ----
__global__ void k_final(const float* __restrict__ x, const float* __restrict__ m0,
                        const float* __restrict__ r0, const float* __restrict__ g0,
                        const float* __restrict__ b0, const float* __restrict__ z2,
                        const float* __restrict__ m2, const float* __restrict__ r2,
                        const float* __restrict__ g2, const float* __restrict__ b2,
                        float* __restrict__ out, int total) {
    int t = blockIdx.x * blockDim.x + threadIdx.x;
    if (t >= total) return;
    int c = t & (C - 1);
    float xn = g0[c] * (x[t] - m0[c]) * r0[c] + b0[c];
    float h  = g2[c] * (z2[t] - m2[c]) * r2[c] + b2[c];
    out[t] = xn + h;
}

extern "C" void kernel_launch(void* const* d_in, const int* in_sizes, int n_in,
                              void* d_out, int out_size, void* d_ws, size_t ws_size,
                              hipStream_t stream) {
    const float*     x    = (const float*)d_in[0];
    const long long* edge = (const long long*)d_in[1];   // int64 [2, E]
    const float* g0 = (const float*)d_in[2];
    const float* be0 = (const float*)d_in[3];
    const float* W1 = (const float*)d_in[4];
    const float* b1 = (const float*)d_in[5];
    const float* g1 = (const float*)d_in[6];
    const float* be1 = (const float*)d_in[7];
    const float* W2 = (const float*)d_in[8];
    const float* b2 = (const float*)d_in[9];
    const float* g2 = (const float*)d_in[10];
    const float* be2 = (const float*)d_in[11];
    float* out = (float*)d_out;

    const int n = in_sizes[0] / C;
    const int E = in_sizes[1] / 2;
    const long long* src = edge;
    const long long* dst = edge + E;

    // workspace carve-out (256B aligned)
    char* ws = (char*)d_ws;
    size_t off = 0;
    auto carve = [&](size_t bytes) -> void* {
        void* p = ws + off;
        off = (off + bytes + 255) & ~(size_t)255;
        return p;
    };
    _Float16* h16  = (_Float16*)carve((size_t)n * C * 2);   // f16 GEMM operand
    float*    hgem = (float*)   carve((size_t)n * C * 4);   // GEMM output
    float*    agg  = (float*)   carve((size_t)n * C * 4);   // aggregation / z (in place)
    float*    dinv = (float*)   carve((size_t)n * 4);       // deg -> rsqrt(deg)
    _Float16* W1h  = (_Float16*)carve(C * C * 2);
    _Float16* W2h  = (_Float16*)carve(C * C * 2);
    float*    st   = (float*)   carve(768 * 4);             // 3 BN x (sum|mean, sumsq|rstd)
    float *s0 = st,       *q0 = st + 128;
    float *s1 = st + 256, *q1 = st + 384;
    float *s2 = st + 512, *q2 = st + 640;

    const int total = n * C;
    const int gElem = (total + 255) / 256;
    const int RPB = 256;
    const int gCol = (n + RPB - 1) / RPB;
    const int TPB = 10;
    const int gGemm = ((n + 15) / 16 + TPB - 1) / TPB;
    const long long eTotal = (long long)E * C;
    const int gEdge = (int)((eTotal + 255) / 256);

    // degree + BN0 stats
    k_init<<<(n + 255) / 256, 256, 0, stream>>>(dinv, st, n);
    k_deg<<<(E + 255) / 256, 256, 0, stream>>>(dst, dinv, E);
    k_dinv<<<(n + 255) / 256, 256, 0, stream>>>(dinv, n);
    k_colstats<<<gCol, 128, 0, stream>>>(x, s0, q0, n, RPB);
    k_finalize<<<1, 128, 0, stream>>>(s0, q0, (float)n);
    k_bn_apply_f16<<<gElem, 256, 0, stream>>>(x, s0, q0, g0, be0, h16, total);
    k_f32_to_f16<<<64, 256, 0, stream>>>(W1, W1h, C * C);
    k_f32_to_f16<<<64, 256, 0, stream>>>(W2, W2h, C * C);

    // layer 1 (GEMM writes h and agg = h*dinv^2 in one pass)
    k_gemm_wmma<<<gGemm, 256, 0, stream>>>(h16, W1h, hgem, agg, dinv, n, TPB);
    k_edge<<<gEdge, 256, 0, stream>>>(src, dst, hgem, dinv, agg, eTotal);
    k_epi_stats<<<gCol, 128, 0, stream>>>(agg, b1, s1, q1, n, RPB);
    k_finalize<<<1, 128, 0, stream>>>(s1, q1, (float)n);
    k_bn_apply_f16<<<gElem, 256, 0, stream>>>(agg, s1, q1, g1, be1, h16, total);

    // layer 2
    k_gemm_wmma<<<gGemm, 256, 0, stream>>>(h16, W2h, hgem, agg, dinv, n, TPB);
    k_edge<<<gEdge, 256, 0, stream>>>(src, dst, hgem, dinv, agg, eTotal);
    k_epi_stats<<<gCol, 128, 0, stream>>>(agg, b2, s2, q2, n, RPB);
    k_finalize<<<1, 128, 0, stream>>>(s2, q2, (float)n);

    // residual + BN2
    k_final<<<gElem, 256, 0, stream>>>(x, s0, q0, g0, be0, agg, s2, q2, g2, be2, out, total);
}
